// LSTM_70282844831914
// MI455X (gfx1250) — matrix operand order repaired and checked by reading, not compiled, not run on hardware
//
#include <hip/hip_runtime.h>
#include <hip/hip_bf16.h>

typedef __attribute__((ext_vector_type(16))) __bf16    v16bf;
typedef __attribute__((ext_vector_type(8)))  float     v8f;
typedef __attribute__((ext_vector_type(4)))  unsigned  u32x4;

#define S_LEN   1024
#define BATCH   128
#define IN_DIM  256
#define HID     512
#define L1_DIM  256
#define O_DIM   64
#define KSTEPS_X 8          // 256 / 32
#define KSTEPS_H 16         // 512 / 32
#define KSTEPS   24
#define NWGC    32          // column groups: 512 / 16
#define NWGR    4           // row groups: 128 / 32
#define NWG_TOT (NWGC * NWGR)
#define FRAG_ELEMS 512      // 32 lanes * 16 bf16 per B fragment
#define WG_B_ELEMS (KSTEPS * 4 * FRAG_ELEMS)   // 49152 bf16 = 96 KB per WG
#define WG_B_VEC   (WG_B_ELEMS / 8)            // 6144 x 16B chunks

union Frag {
    u32x4  u[2];
    v16bf  v;
    __bf16 h[16];
};

// ---------------------------------------------------------------- init ----
__global__ void lstm_init(unsigned* __restrict__ cnt,
                          __bf16*   __restrict__ Hbf0,
                          float*    __restrict__ Cst) {
    int t = blockIdx.x * blockDim.x + threadIdx.x;   // 65536 threads
    if (t < S_LEN) cnt[t] = 0u;
    if (t < BATCH * HID) {
        Hbf0[t] = (__bf16)0.0f;
        Cst[t]  = 0.0f;
    }
}

// --------------------------------------------- X: [B,S,I] f32 -> [S,B,I] bf16
__global__ void lstm_convert_x(const float* __restrict__ X,
                               __bf16* __restrict__ Xbf) {
    size_t t  = (size_t)blockIdx.x * blockDim.x + threadIdx.x;
    int    i  = (int)(t % IN_DIM);
    size_t sb = t / IN_DIM;
    int    b  = (int)(sb % BATCH);
    int    s  = (int)(sb / BATCH);
    Xbf[t] = (__bf16)X[((size_t)b * S_LEN + s) * IN_DIM + i];
}

// --------- pack stacked weights [Wx;Wh] (768 x 512, per gate) into WMMA-B
// fragment order: ((cg*24 + ks)*4 + gate), each fragment = 32 lanes x 32B
__global__ void lstm_pack_w(const float* __restrict__ Wxi, const float* __restrict__ Wxf,
                            const float* __restrict__ Wxo, const float* __restrict__ Wxc,
                            const float* __restrict__ Whi, const float* __restrict__ Whf,
                            const float* __restrict__ Who, const float* __restrict__ Whc,
                            __bf16* __restrict__ Bpack) {
    int tid  = blockIdx.x * blockDim.x + threadIdx.x;   // 98304 threads
    int lane = tid & 31;
    int frag = tid >> 5;                 // 0..3071
    int g    = frag & 3;
    int ks   = (frag >> 2) % KSTEPS;
    int cg   = frag / (KSTEPS * 4);
    const float* Wx = (g == 0) ? Wxi : (g == 1) ? Wxf : (g == 2) ? Wxo : Wxc;
    const float* Wh = (g == 0) ? Whi : (g == 1) ? Whf : (g == 2) ? Who : Whc;
    int colg  = cg * 16 + (lane & 15);
    int kbase = ks * 32 + ((lane & 16) ? 8 : 0);    // lanes 0-15: K 0..7,16..23; 16-31: K 8..15,24..31
    Frag f;
#pragma unroll
    for (int j = 0; j < 16; ++j) {
        int   k = kbase + j + ((j >= 8) ? 8 : 0);
        float v = (k < IN_DIM) ? Wx[(size_t)k * HID + colg]
                               : Wh[(size_t)(k - IN_DIM) * HID + colg];
        f.h[j] = (__bf16)v;
    }
    u32x4* dst = (u32x4*)(Bpack + (size_t)frag * FRAG_ELEMS + lane * 16);
    dst[0] = f.u[0];
    dst[1] = f.u[1];
}

// ---------------------------------------------------------------- scan ----
__device__ __forceinline__ float sigm_f(float x) {
    x = fminf(fmaxf(x, -30.0f), 30.0f);
    return 1.0f / (1.0f + __expf(-x));
}
__device__ __forceinline__ float tanh_f(float x) {
    x = fminf(fmaxf(x, -15.0f), 15.0f);
    float e = __expf(2.0f * x);
    return (e - 1.0f) / (e + 1.0f);
}

__device__ __forceinline__ v16bf load_frag16(const __bf16* p) {
    Frag f;
    f.u[0] = *(const u32x4*)(p);
    f.u[1] = *(const u32x4*)(p + 8);
    return f.v;
}

__global__ void lstm_scan(const __bf16* __restrict__ Xbf,
                          const __bf16* __restrict__ Bpack,
                          const float* __restrict__ b_i, const float* __restrict__ b_f,
                          const float* __restrict__ b_o, const float* __restrict__ b_c,
                          __bf16* __restrict__ Hbf0, __bf16* __restrict__ Hbf1,
                          float* __restrict__ Cst,
                          float* __restrict__ outHf, float* __restrict__ outCf,
                          unsigned* __restrict__ cnt) {
    extern __shared__ __bf16 sB[];          // 96 KB: this WG's weight fragments

    const int cg   = blockIdx.x;            // 0..31  : 16 H-columns each
    const int rg   = blockIdx.y;            // 0..3   : 32 batch rows each
    const int wave = threadIdx.x >> 5;      // 2 waves per WG, one M-tile each
    const int lane = threadIdx.x & 31;
    const int rbase = rg * 32 + wave * 16;
    const int colL  = lane & 15;
    const int col   = cg * 16 + colL;
    const int arow  = rbase + colL;                 // A-matrix row for this lane
    const int hi16  = (lane >> 4) & 1;
    const int aoff  = hi16 * 8;                     // A layout K sub-offset
    const int rowtop = rbase + hi16 * 8;            // C/D layout row base

    // ---- stage this column-group's 96 KB of packed weights into LDS ----
    {
        const u32x4* src = (const u32x4*)(Bpack + (size_t)cg * WG_B_ELEMS);
        u32x4*       dst = (u32x4*)sB;
        for (int i = threadIdx.x; i < WG_B_VEC; i += blockDim.x)
            dst[i] = src[i];
    }
    __syncthreads();

    const float bias0 = b_i[col], bias1 = b_f[col], bias2 = b_o[col], bias3 = b_c[col];
    const __bf16* bLane = sB + lane * 16;   // per-lane base inside each fragment
    __bf16* hbuf[2] = { Hbf0, Hbf1 };

    // laundered zero offset: defeats LICM hoisting (and re-spilling) of the
    // loop-invariant LDS weight loads while preserving addrspace(3) inference
    unsigned lzero = 0;

    for (int t = 0; t < S_LEN; ++t) {
        asm volatile("" : "+v"(lzero));     // opaque each iteration, no code
        const __bf16* bl = bLane + lzero;

        const __bf16* Hread  = hbuf[t & 1];
        __bf16*       Hwrite = hbuf[(t + 1) & 1];
        const __bf16* xr = Xbf + ((size_t)t * BATCH + arow) * IN_DIM;
        const __bf16* hr = Hread + (size_t)arow * HID;

        v8f acc0 = {}, acc1 = {}, acc2 = {}, acc3 = {};

        // ---- input-projection part of K (X_t columns 0..255) ----
        for (int ks = 0; ks < KSTEPS_X; ++ks) {
            Frag fa;
            const __bf16* ap = xr + ks * 32 + aoff;
            fa.u[0] = *(const u32x4*)(ap);
            fa.u[1] = *(const u32x4*)(ap + 16);
            const __bf16* bp = bl + (size_t)ks * (4 * FRAG_ELEMS);
            v16bf b0 = load_frag16(bp);
            v16bf b1 = load_frag16(bp + 1 * FRAG_ELEMS);
            v16bf b2 = load_frag16(bp + 2 * FRAG_ELEMS);
            v16bf b3 = load_frag16(bp + 3 * FRAG_ELEMS);
            acc0 = __builtin_amdgcn_wmma_f32_16x16x32_bf16(false, fa.v, false, b0, (short)0, acc0, false, false);
            acc1 = __builtin_amdgcn_wmma_f32_16x16x32_bf16(false, fa.v, false, b1, (short)0, acc1, false, false);
            acc2 = __builtin_amdgcn_wmma_f32_16x16x32_bf16(false, fa.v, false, b2, (short)0, acc2, false, false);
            acc3 = __builtin_amdgcn_wmma_f32_16x16x32_bf16(false, fa.v, false, b3, (short)0, acc3, false, false);
        }
        // ---- recurrent part of K (H_{t-1} columns 0..511) ----
        for (int ks = 0; ks < KSTEPS_H; ++ks) {
            Frag fa;
            const __bf16* ap = hr + ks * 32 + aoff;
            fa.u[0] = *(const u32x4*)(ap);
            fa.u[1] = *(const u32x4*)(ap + 16);
            const __bf16* bp = bl + (size_t)(KSTEPS_X + ks) * (4 * FRAG_ELEMS);
            v16bf b0 = load_frag16(bp);
            v16bf b1 = load_frag16(bp + 1 * FRAG_ELEMS);
            v16bf b2 = load_frag16(bp + 2 * FRAG_ELEMS);
            v16bf b3 = load_frag16(bp + 3 * FRAG_ELEMS);
            acc0 = __builtin_amdgcn_wmma_f32_16x16x32_bf16(false, fa.v, false, b0, (short)0, acc0, false, false);
            acc1 = __builtin_amdgcn_wmma_f32_16x16x32_bf16(false, fa.v, false, b1, (short)0, acc1, false, false);
            acc2 = __builtin_amdgcn_wmma_f32_16x16x32_bf16(false, fa.v, false, b2, (short)0, acc2, false, false);
            acc3 = __builtin_amdgcn_wmma_f32_16x16x32_bf16(false, fa.v, false, b3, (short)0, acc3, false, false);
        }

        // ---- gate math (C/D layout: VGPR r -> row rowtop+r, col = lane%16) ----
#pragma unroll
        for (int r = 0; r < 8; ++r) {
            int   row = rowtop + r;
            float Ig = sigm_f(acc0[r] + bias0);
            float Fg = sigm_f(acc1[r] + bias1);
            float Og = sigm_f(acc2[r] + bias2);
            float Ct = tanh_f(acc3[r] + bias3);
            size_t idx = (size_t)row * HID + col;
            float c  = Cst[idx];
            float cn = Fg * c + Ig * Ct;
            float hn = tanh_f(cn) + Og;          // reference: tanh(C) + O
            Cst[idx]    = cn;
            Hwrite[idx] = (__bf16)hn;
            if (t == S_LEN - 1) {
                outHf[idx] = hn;
                outCf[idx] = cn;
            }
        }

        // ---- grid-wide barrier ----
        __syncthreads();
        __threadfence();
        if (threadIdx.x == 0) {
            __hip_atomic_fetch_add(&cnt[t], 1u, __ATOMIC_ACQ_REL, __HIP_MEMORY_SCOPE_AGENT);
            while (__hip_atomic_load(&cnt[t], __ATOMIC_ACQUIRE, __HIP_MEMORY_SCOPE_AGENT) < NWG_TOT) {
                __builtin_amdgcn_s_sleep(1);
            }
        }
        __syncthreads();
    }
}

// ---------------------------------------------------------------- head ----
__global__ void lstm_head(const float* __restrict__ Hf,
                          const float* __restrict__ Wo1, const float* __restrict__ bo1,
                          const float* __restrict__ Wo2, const float* __restrict__ bo2,
                          float* __restrict__ out) {
    __shared__ float sH[HID];
    __shared__ float sh1[L1_DIM];
    __shared__ float so[O_DIM];
    __shared__ float smax, ssum;
    const int row = blockIdx.x;
    const int tid = threadIdx.x;

    for (int k = tid; k < HID; k += blockDim.x) sH[k] = Hf[(size_t)row * HID + k];
    __syncthreads();

    if (tid < L1_DIM) {
        float acc = bo1[tid];
        for (int k = 0; k < HID; ++k) acc += sH[k] * Wo1[(size_t)k * L1_DIM + tid];
        sh1[tid] = fmaxf(acc, 0.0f);
    }
    __syncthreads();

    if (tid < O_DIM) {
        float acc = bo2[tid];
        for (int k = 0; k < L1_DIM; ++k) acc += sh1[k] * Wo2[(size_t)k * O_DIM + tid];
        so[tid] = acc;
    }
    __syncthreads();

    if (tid == 0) {
        float m = so[0];
        for (int o = 1; o < O_DIM; ++o) m = fmaxf(m, so[o]);
        smax = m;
    }
    __syncthreads();
    if (tid < O_DIM) so[tid] = __expf(so[tid] - smax);
    __syncthreads();
    if (tid == 0) {
        float s = 0.0f;
        for (int o = 0; o < O_DIM; ++o) s += so[o];
        ssum = s;
    }
    __syncthreads();
    if (tid < O_DIM) out[(size_t)row * O_DIM + tid] = so[tid] / ssum;
}

// -------------------------------------------------------------- launch ----
extern "C" void kernel_launch(void* const* d_in, const int* in_sizes, int n_in,
                              void* d_out, int out_size, void* d_ws, size_t ws_size,
                              hipStream_t stream) {
    const float* X   = (const float*)d_in[0];
    const float* Wxi = (const float*)d_in[1];
    const float* Whi = (const float*)d_in[2];
    const float* bi  = (const float*)d_in[3];
    const float* Wxf = (const float*)d_in[4];
    const float* Whf = (const float*)d_in[5];
    const float* bf_ = (const float*)d_in[6];
    const float* Wxo = (const float*)d_in[7];
    const float* Who = (const float*)d_in[8];
    const float* bo  = (const float*)d_in[9];
    const float* Wxc = (const float*)d_in[10];
    const float* Whc = (const float*)d_in[11];
    const float* bc  = (const float*)d_in[12];
    const float* Wo1 = (const float*)d_in[13];
    const float* bo1 = (const float*)d_in[14];
    const float* Wo2 = (const float*)d_in[15];
    const float* bo2 = (const float*)d_in[16];

    char*  ws  = (char*)d_ws;
    size_t off = 0;
    unsigned* cnt = (unsigned*)(ws);                    off += 8192;
    __bf16* Xbf   = (__bf16*)(ws + off);                off += (size_t)S_LEN * BATCH * IN_DIM * 2;
    __bf16* Bpack = (__bf16*)(ws + off);                off += (size_t)NWGC * WG_B_ELEMS * 2;
    __bf16* Hbf0  = (__bf16*)(ws + off);                off += (size_t)BATCH * HID * 2;
    __bf16* Hbf1  = (__bf16*)(ws + off);                off += (size_t)BATCH * HID * 2;
    float*  Cst   = (float*)(ws + off);                 off += (size_t)BATCH * HID * 4;

    float* out   = (float*)d_out;
    float* outHf = out + BATCH * O_DIM;
    float* outCf = outHf + BATCH * HID;

    lstm_init<<<256, 256, 0, stream>>>(cnt, Hbf0, Cst);
    lstm_convert_x<<<(S_LEN * BATCH * IN_DIM) / 256, 256, 0, stream>>>(X, Xbf);
    lstm_pack_w<<<(NWGC * KSTEPS * 4 * 32) / 256, 256, 0, stream>>>(
        Wxi, Wxf, Wxo, Wxc, Whi, Whf, Who, Whc, Bpack);
    lstm_scan<<<dim3(NWGC, NWGR), 64, (size_t)WG_B_ELEMS * 2, stream>>>(
        Xbf, Bpack, bi, bf_, bo, bc, Hbf0, Hbf1, Cst, outHf, outCf, cnt);
    lstm_head<<<BATCH, 256, 0, stream>>>(outHf, Wo1, bo1, Wo2, bo2, out);
}
